// PSCSNetwork_69664369541721
// MI455X (gfx1250) — matrix operand order repaired and checked by reading, not compiled, not run on hardware
//
#include <hip/hip_runtime.h>
#include <hip/hip_bf16.h>

// ---------------------------------------------------------------------------
// PSCSNetwork (code2vec-style) for MI455X / gfx1250, wave32.
//   B=128, S=30, K=200, P=20, E=128.  Dominant cost: bi-LSTM GEMMs (~268 GF)
//   -> fused per-step bf16 WMMA kernel (v_wmma_f32_16x16x32_bf16) with
//      async Global->LDS double-buffered x tiles (ASYNCcnt path) and all
//      weight B-fragments hoisted into the 1024-VGPR register file.
// ---------------------------------------------------------------------------

#define BB   128
#define SS   30
#define KK   200
#define PP   20
#define EE   128
#define NROW (BB * KK)        // 25600 path rows
#define TT   PP               // 20 timesteps

typedef __attribute__((ext_vector_type(16))) __bf16 v16bf;
typedef __attribute__((ext_vector_type(8)))  __bf16 v8bf;
typedef __attribute__((ext_vector_type(8)))  float  v8f;

__device__ __forceinline__ float sigmoidf_(float x) {
    return 1.0f / (1.0f + __expf(-x));
}

// Async copy 16B Global -> LDS (gfx1250, tracked by ASYNCcnt).
__device__ __forceinline__ void async_load_b128(unsigned lds_addr, const void* gaddr) {
    asm volatile("global_load_async_to_lds_b128 %0, %1, off"
                 :: "v"(lds_addr), "v"(gaddr) : "memory");
}
__device__ __forceinline__ void wait_asynccnt0() {
    asm volatile("s_wait_asynccnt 0x0" ::: "memory");
}

// -------------------------------------------------------------------------
// Convert the 4 LSTM weight matrices (each 512x128 f32) to bf16 in ws.
// -------------------------------------------------------------------------
__global__ __launch_bounds__(256)
void cvt_w_kernel(const float* __restrict__ a, const float* __restrict__ b,
                  const float* __restrict__ c, const float* __restrict__ d,
                  __bf16* __restrict__ out) {
    int i = blockIdx.x * 256 + threadIdx.x;      // 0 .. 262143
    int m = i >> 16;
    int j = i & 65535;
    const float* src = (m == 0) ? a : (m == 1) ? b : (m == 2) ? c : d;
    out[(size_t)m * 65536 + j] = (__bf16)src[j];
}

// -------------------------------------------------------------------------
// Embedding gather with max_norm=1.0 clipping. One wave per output row,
// 4 floats per lane, cross-lane L2-norm via shfl_xor (wave32).
// -------------------------------------------------------------------------
__global__ __launch_bounds__(256)
void embed_rows_f32_kernel(const float* __restrict__ table, const int* __restrict__ idx,
                           float* __restrict__ out, int nrows) {
    int row  = blockIdx.x * 8 + (threadIdx.x >> 5);
    int lane = threadIdx.x & 31;
    if (row >= nrows) return;
    int t = idx[row];
    float4 v = ((const float4*)(table + (size_t)t * EE))[lane];
    float ss = v.x * v.x + v.y * v.y + v.z * v.z + v.w * v.w;
    for (int m = 16; m >= 1; m >>= 1) ss += __shfl_xor(ss, m, 32);
    float n  = sqrtf(ss);
    float sc = (n > 1.0f) ? 1.0f / fmaxf(n, 1e-12f) : 1.0f;
    ((float4*)(out + (size_t)row * EE))[lane] = make_float4(v.x*sc, v.y*sc, v.z*sc, v.w*sc);
}

__global__ __launch_bounds__(256)
void embed_rows_bf16_kernel(const float* __restrict__ table, const int* __restrict__ idx,
                            __bf16* __restrict__ out, int nrows) {
    int row  = blockIdx.x * 8 + (threadIdx.x >> 5);
    int lane = threadIdx.x & 31;
    if (row >= nrows) return;
    int t = idx[row];
    float4 v = ((const float4*)(table + (size_t)t * EE))[lane];
    float ss = v.x * v.x + v.y * v.y + v.z * v.z + v.w * v.w;
    for (int m = 16; m >= 1; m >>= 1) ss += __shfl_xor(ss, m, 32);
    float n  = sqrtf(ss);
    float sc = (n > 1.0f) ? 1.0f / fmaxf(n, 1e-12f) : 1.0f;
    __bf16* po = out + (size_t)row * EE + lane * 4;
    po[0] = (__bf16)(v.x * sc);
    po[1] = (__bf16)(v.y * sc);
    po[2] = (__bf16)(v.z * sc);
    po[3] = (__bf16)(v.w * sc);
}

// start/end token embeddings: gather S tokens per row, norm-clip, sum.
__global__ __launch_bounds__(256)
void embed_sum_kernel(const float* __restrict__ table, const int* __restrict__ idx,
                      float* __restrict__ out, int nrows, int seq) {
    int row  = blockIdx.x * 8 + (threadIdx.x >> 5);
    int lane = threadIdx.x & 31;
    if (row >= nrows) return;
    float ax = 0.f, ay = 0.f, az = 0.f, aw = 0.f;
    for (int s = 0; s < seq; ++s) {
        int t = idx[row * seq + s];
        float4 v = ((const float4*)(table + (size_t)t * EE))[lane];
        float ss = v.x * v.x + v.y * v.y + v.z * v.z + v.w * v.w;
        for (int m = 16; m >= 1; m >>= 1) ss += __shfl_xor(ss, m, 32);
        float n  = sqrtf(ss);
        float sc = (n > 1.0f) ? 1.0f / fmaxf(n, 1e-12f) : 1.0f;
        ax += v.x * sc; ay += v.y * sc; az += v.z * sc; aw += v.w * sc;
    }
    ((float4*)(out + (size_t)row * EE))[lane] = make_float4(ax, ay, az, aw);
}

// -------------------------------------------------------------------------
// Fused bi-LSTM over T=20 steps. Block = 256 threads (8 wave32) owns 32 rows.
// Wave w owns hidden units [16w,16w+16): computes all 4 gates for them, so
// the i/f/g/o combine is wave-local; c stays in VGPRs, h round-trips LDS as
// bf16. x tiles are staged Global->LDS with async loads, double-buffered so
// the fetch of step s+1 overlaps the 64 WMMAs of step s. All Wih/Whh
// B-fragments (256 VGPRs) are loaded ONCE and held in registers across all
// 20 steps (CDNA5 1024-VGPR file, s_set_vgpr_msb).
// grid = (NROW/32, 2): blockIdx.y selects forward/backward direction.
// -------------------------------------------------------------------------
__global__ __launch_bounds__(256)
void lstm_kernel(const __bf16* __restrict__ pe,     // [NROW][TT][EE] bf16
                 const __bf16* __restrict__ wih_f, const __bf16* __restrict__ whh_f,
                 const float*  __restrict__ bih_f, const float*  __restrict__ bhh_f,
                 const __bf16* __restrict__ wih_b, const __bf16* __restrict__ whh_b,
                 const float*  __restrict__ bih_b, const float*  __restrict__ bhh_b,
                 float* __restrict__ hf, float* __restrict__ hb) {
    const int dir = blockIdx.y;
    const __bf16* Wih = dir ? wih_b : wih_f;
    const __bf16* Whh = dir ? whh_b : whh_f;
    const float*  Bih = dir ? bih_b : bih_f;
    const float*  Bhh = dir ? bhh_b : bhh_f;
    float* Hout = dir ? hb : hf;

    const int tid  = threadIdx.x;
    const int w    = tid >> 5;
    const int lane = tid & 31;
    const int ln   = lane & 15;         // N within 16-wide tile
    const int nh   = lane >> 4;         // half-lane group
    const int hid  = w * 16 + ln;       // hidden unit owned by this lane
    const int row0 = blockIdx.x * 32;

    __shared__ __bf16 xlds[2 * 32 * EE];  // double-buffered x tile, 2 x 8 KB
    __shared__ __bf16 hlds[32 * EE];      // current h, bf16, row-major [32][128]

    const unsigned xbase = (unsigned)(size_t)(&xlds[0]);  // LDS byte address

    for (int i = tid; i < 32 * EE; i += 256) hlds[i] = (__bf16)0.0f;

    // Preload x tile for step 0 into buffer 0 (async Global->LDS).
    {
        const int t0 = dir ? (TT - 1) : 0;
#pragma unroll
        for (int q = 0; q < 2; ++q) {
            const int chunk = q * 4096 + tid * 16;    // byte offset in 8 KB tile
            const int r     = chunk >> 8;             // 256 B per row
            const int cb    = chunk & 255;
            const void* g = (const char*)(pe + ((size_t)(row0 + r) * TT + t0) * EE) + cb;
            async_load_b128(xbase + chunk, g);
        }
        wait_asynccnt0();
    }

    // Hoist ALL weight B-fragments into registers: loop-invariant across the
    // 20 timesteps. 4 kt x 4 gates x 2 matrices x 8 VGPRs = 256 VGPRs.
    v16bf wih_frag[4][4], whh_frag[4][4];
#pragma unroll
    for (int kt = 0; kt < 4; ++kt) {
        const int kbB = kt * 32 + nh * 16;
#pragma unroll
        for (int g = 0; g < 4; ++g) {
            const int col = g * EE + hid;  // gate column (row of W, since B = W^T)
            wih_frag[kt][g] = *(const v16bf*)(Wih + (size_t)col * EE + kbB);
            whh_frag[kt][g] = *(const v16bf*)(Whh + (size_t)col * EE + kbB);
        }
    }

    float bias[4];
#pragma unroll
    for (int g = 0; g < 4; ++g) bias[g] = Bih[g * EE + hid] + Bhh[g * EE + hid];

    float c[2][8];
    float hreg[2][8];
#pragma unroll
    for (int m = 0; m < 2; ++m)
#pragma unroll
        for (int j = 0; j < 8; ++j) { c[m][j] = 0.0f; hreg[m][j] = 0.0f; }

    __syncthreads();

    int buf = 0;
    for (int s = 0; s < TT; ++s) {
        // Kick off async load of next step's x tile into the other buffer.
        if (s + 1 < TT) {
            const int tn = dir ? (TT - 2 - s) : (s + 1);
            const int nb = buf ^ 1;
#pragma unroll
            for (int q = 0; q < 2; ++q) {
                const int chunk = q * 4096 + tid * 16;
                const int r     = chunk >> 8;
                const int cb    = chunk & 255;
                const void* g = (const char*)(pe + ((size_t)(row0 + r) * TT + tn) * EE) + cb;
                async_load_b128(xbase + nb * 8192 + chunk, g);
            }
        }

        v8f acc[2][4];
#pragma unroll
        for (int m = 0; m < 2; ++m)
#pragma unroll
            for (int g = 0; g < 4; ++g)
#pragma unroll
                for (int j = 0; j < 8; ++j) acc[m][g][j] = bias[g];

#pragma unroll
        for (int kt = 0; kt < 4; ++kt) {
            const int k0  = kt * 32;
            const int kbA = k0 + nh * 16;      // A-fragment K base for this lane
            v16bf ax[2], ah[2];
#pragma unroll
            for (int m = 0; m < 2; ++m) {
                const __bf16* px = xlds + buf * (32 * EE) + (m * 16 + ln) * EE;
                v8bf xlo = *(const v8bf*)(px + kbA);
                v8bf xhi = *(const v8bf*)(px + kbA + 8);
                const __bf16* ph = hlds + (m * 16 + ln) * EE;
                v8bf hlo = *(const v8bf*)(ph + kbA);
                v8bf hhi = *(const v8bf*)(ph + kbA + 8);
#pragma unroll
                for (int i = 0; i < 8; ++i) {
                    ax[m][i] = xlo[i]; ax[m][i + 8] = xhi[i];
                    ah[m][i] = hlo[i]; ah[m][i + 8] = hhi[i];
                }
            }
#pragma unroll
            for (int g = 0; g < 4; ++g) {
#pragma unroll
                for (int m = 0; m < 2; ++m) {
                    acc[m][g] = __builtin_amdgcn_wmma_f32_16x16x32_bf16(
                        false, ax[m], false, wih_frag[kt][g], (short)0, acc[m][g], false, false);
                    acc[m][g] = __builtin_amdgcn_wmma_f32_16x16x32_bf16(
                        false, ah[m], false, whh_frag[kt][g], (short)0, acc[m][g], false, false);
                }
            }
        }

        // Gate combine (torch order i,f,g,o), wave-local.
#pragma unroll
        for (int m = 0; m < 2; ++m)
#pragma unroll
            for (int j = 0; j < 8; ++j) {
                float iv = acc[m][0][j];
                float fv = acc[m][1][j];
                float gv = acc[m][2][j];
                float ov = acc[m][3][j];
                float cn = sigmoidf_(fv) * c[m][j] + sigmoidf_(iv) * tanhf(gv);
                float hn = sigmoidf_(ov) * tanhf(cn);
                c[m][j]    = cn;
                hreg[m][j] = hn;
            }

        __syncthreads();   // everyone done reading previous h / current x
#pragma unroll
        for (int m = 0; m < 2; ++m)
#pragma unroll
            for (int j = 0; j < 8; ++j)
                hlds[(m * 16 + j + nh * 8) * EE + hid] = (__bf16)hreg[m][j];

        wait_asynccnt0();  // this wave's async x loads for s+1 are in LDS
        __syncthreads();   // new h + next x tile visible to all waves
        buf ^= 1;
    }

    // Final hidden state, f32.
#pragma unroll
    for (int m = 0; m < 2; ++m)
#pragma unroll
        for (int j = 0; j < 8; ++j)
            Hout[(size_t)(row0 + m * 16 + j + nh * 8) * EE + hid] = hreg[m][j];
}

// -------------------------------------------------------------------------
// NL attention: per-b block. x=[30,128], scores = x . (W_b @ mean(x)).
// -------------------------------------------------------------------------
__global__ __launch_bounds__(256)
void attn_nl_kernel(const float* __restrict__ x,   // [B,30,128]
                    const float* __restrict__ Wb,  // [128,128]
                    float* __restrict__ out) {     // [B,128]
    const int b   = blockIdx.x;
    const int tid = threadIdx.x;
    __shared__ float h0[EE], v[EE], sc[32];
    const float* xb = x + (size_t)b * SS * EE;

    if (tid < EE) {
        float s = 0.f;
        for (int l = 0; l < SS; ++l) s += xb[l * EE + tid];
        h0[tid] = s * (1.0f / SS);
    }
    __syncthreads();
    if (tid < EE) {
        float s = 0.f;
        for (int e = 0; e < EE; ++e) s += Wb[tid * EE + e] * h0[e];
        v[tid] = s;
    }
    __syncthreads();
    const int w = tid >> 5, lane = tid & 31;
    for (int l = w; l < SS; l += 8) {
        float p = 0.f;
        for (int i = 0; i < 4; ++i) {
            int d = lane * 4 + i;
            p += xb[l * EE + d] * v[d];
        }
        for (int m = 16; m >= 1; m >>= 1) p += __shfl_xor(p, m, 32);
        if (lane == 0) sc[l] = p;
    }
    __syncthreads();
    if (tid == 0) {
        float mx = -1e30f;
        for (int l = 0; l < SS; ++l) mx = fmaxf(mx, sc[l]);
        float sum = 0.f;
        for (int l = 0; l < SS; ++l) { float e = __expf(sc[l] - mx); sc[l] = e; sum += e; }
        float inv = 1.0f / sum;
        for (int l = 0; l < SS; ++l) sc[l] *= inv;
    }
    __syncthreads();
    if (tid < EE) {
        float s = 0.f;
        for (int l = 0; l < SS; ++l) s += sc[l] * xb[l * EE + tid];
        out[(size_t)b * EE + tid] = s;
    }
}

// -------------------------------------------------------------------------
// Code attention + final linear. code_emb[b,k,:] = [st|en|h_f|h_b] (virtual).
// -------------------------------------------------------------------------
__device__ __forceinline__ float ce_val(const float* st, const float* en,
                                        const float* hf, const float* hb,
                                        int b, int k, int d) {
    const size_t r = (size_t)b * KK + k;
    if (d < 128)       return st[r * EE + d];
    else if (d < 256)  return en[r * EE + (d - 128)];
    else if (d < 384)  return hf[r * EE + (d - 256)];
    else               return hb[r * EE + (d - 384)];
}

__global__ __launch_bounds__(256)
void attn_code_kernel(const float* __restrict__ st, const float* __restrict__ en,
                      const float* __restrict__ hf, const float* __restrict__ hb,
                      const float* __restrict__ Wa,    // [512,512]
                      const float* __restrict__ linW,  // [128,512]
                      const float* __restrict__ linb,  // [128]
                      float* __restrict__ out) {       // [B,128]
    const int b   = blockIdx.x;
    const int tid = threadIdx.x;
    __shared__ float h0[512], v[512], sc[KK], ov[512];

    for (int d = tid; d < 512; d += 256) {
        float s = 0.f;
        for (int k = 0; k < KK; ++k) s += ce_val(st, en, hf, hb, b, k, d);
        h0[d] = s * (1.0f / KK);
    }
    __syncthreads();
    for (int d = tid; d < 512; d += 256) {
        float s = 0.f;
        for (int e = 0; e < 512; ++e) s += Wa[(size_t)d * 512 + e] * h0[e];
        v[d] = s;
    }
    __syncthreads();
    const int w = tid >> 5, lane = tid & 31;
    for (int k = w; k < KK; k += 8) {
        float p = 0.f;
        for (int i = 0; i < 16; ++i) {
            int d = lane * 16 + i;
            p += ce_val(st, en, hf, hb, b, k, d) * v[d];
        }
        for (int m = 16; m >= 1; m >>= 1) p += __shfl_xor(p, m, 32);
        if (lane == 0) sc[k] = p;
    }
    __syncthreads();
    if (tid == 0) {
        float mx = -1e30f;
        for (int k = 0; k < KK; ++k) mx = fmaxf(mx, sc[k]);
        float sum = 0.f;
        for (int k = 0; k < KK; ++k) { float e = __expf(sc[k] - mx); sc[k] = e; sum += e; }
        float inv = 1.0f / sum;
        for (int k = 0; k < KK; ++k) sc[k] *= inv;
    }
    __syncthreads();
    for (int d = tid; d < 512; d += 256) {
        float s = 0.f;
        for (int k = 0; k < KK; ++k) s += sc[k] * ce_val(st, en, hf, hb, b, k, d);
        ov[d] = s;
    }
    __syncthreads();
    if (tid < EE) {
        float s = linb[tid];
        for (int d = 0; d < 512; ++d) s += linW[(size_t)tid * 512 + d] * ov[d];
        out[(size_t)b * EE + tid] = s;
    }
}

// -------------------------------------------------------------------------
extern "C" void kernel_launch(void* const* d_in, const int* in_sizes, int n_in,
                              void* d_out, int out_size, void* d_ws, size_t ws_size,
                              hipStream_t stream) {
    const int*   nl        = (const int*)d_in[0];
    const int*   path      = (const int*)d_in[1];
    const int*   start_tok = (const int*)d_in[2];
    const int*   end_tok   = (const int*)d_in[3];
    const float* nl_table  = (const float*)d_in[4];
    const float* path_tab  = (const float*)d_in[5];
    const float* W_a       = (const float*)d_in[6];
    const float* W_b       = (const float*)d_in[7];
    const float* lin_W     = (const float*)d_in[8];
    const float* lin_b     = (const float*)d_in[9];
    const float* w_ih_f    = (const float*)d_in[10];
    const float* w_hh_f    = (const float*)d_in[11];
    const float* b_ih_f    = (const float*)d_in[12];
    const float* b_hh_f    = (const float*)d_in[13];
    const float* w_ih_b    = (const float*)d_in[14];
    const float* w_hh_b    = (const float*)d_in[15];
    const float* b_ih_b    = (const float*)d_in[16];
    const float* b_hh_b    = (const float*)d_in[17];

    float* out = (float*)d_out;          // [0:16384) nl_vec, [16384:32768) code_vec

    // ---- workspace layout (bytes) ----
    char* ws = (char*)d_ws;
    size_t off = 0;
    const size_t SZ_ROWVEC = (size_t)NROW * EE * sizeof(float);      // 13.1 MB
    float* st_buf = (float*)(ws + off); off += SZ_ROWVEC;
    float* en_buf = (float*)(ws + off); off += SZ_ROWVEC;
    float* hf_buf = (float*)(ws + off); off += SZ_ROWVEC;
    float* hb_buf = (float*)(ws + off); off += SZ_ROWVEC;
    float* nl_emb = (float*)(ws + off); off += (size_t)BB * SS * EE * sizeof(float);
    __bf16* wbf   = (__bf16*)(ws + off); off += 4ull * 512 * EE * sizeof(__bf16);
    __bf16* wih_f_bf = wbf;
    __bf16* whh_f_bf = wbf + 1 * 512 * EE;
    __bf16* wih_b_bf = wbf + 2 * 512 * EE;
    __bf16* whh_b_bf = wbf + 3 * 512 * EE;
    __bf16* pe_bf = (__bf16*)(ws + off); off += (size_t)NROW * TT * EE * sizeof(__bf16);
    (void)ws_size; (void)in_sizes; (void)n_in; (void)out_size;

    // 1) weights -> bf16
    cvt_w_kernel<<<1024, 256, 0, stream>>>(w_ih_f, w_hh_f, w_ih_b, w_hh_b, wbf);

    // 2) embeddings
    embed_rows_f32_kernel<<<(BB * SS + 7) / 8, 256, 0, stream>>>(nl_table, nl, nl_emb, BB * SS);
    embed_sum_kernel<<<(NROW + 7) / 8, 256, 0, stream>>>(nl_table, start_tok, st_buf, NROW, SS);
    embed_sum_kernel<<<(NROW + 7) / 8, 256, 0, stream>>>(nl_table, end_tok, en_buf, NROW, SS);
    embed_rows_bf16_kernel<<<(NROW * TT + 7) / 8, 256, 0, stream>>>(path_tab, path, pe_bf, NROW * TT);

    // 3) fused bi-LSTM (WMMA + async Global->LDS + register-resident weights)
    dim3 lgrid(NROW / 32, 2);
    lstm_kernel<<<lgrid, 256, 0, stream>>>(pe_bf,
                                           wih_f_bf, whh_f_bf, b_ih_f, b_hh_f,
                                           wih_b_bf, whh_b_bf, b_ih_b, b_hh_b,
                                           hf_buf, hb_buf);

    // 4) attention heads
    attn_nl_kernel<<<BB, 256, 0, stream>>>(nl_emb, W_b, out);
    attn_code_kernel<<<BB, 256, 0, stream>>>(st_buf, en_buf, hf_buf, hb_buf,
                                             W_a, lin_W, lin_b, out + (size_t)BB * EE);
}
